// ContextQueryAttention_7730941133291
// MI455X (gfx1250) — matrix operand order repaired and checked by reading, not compiled, not run on hardware
//
#include <hip/hip_runtime.h>

// Problem constants (fixed by the reference: b=64, d=128, n=1024, m=128)
#define Bb 64
#define Dd 128
#define Nn 1024
#define Mm 128
#define BIGF 1.0e30f

#define GLOBAL_AS __attribute__((address_space(1)))
#define LDS_AS    __attribute__((address_space(3)))

typedef __attribute__((ext_vector_type(16))) __bf16 v16bf;
typedef __attribute__((ext_vector_type(8)))  __bf16 v8bf;
typedef __attribute__((ext_vector_type(4)))  __bf16 v4bf;
typedef __attribute__((ext_vector_type(8)))  float  v8f;
typedef __attribute__((ext_vector_type(4)))  float  v4f;
typedef __attribute__((ext_vector_type(4)))  int    v4i;

__device__ __forceinline__ __bf16 to_bf(float x) { return (__bf16)x; }

// ---- WMMA fragment gathers as 2x 16-byte LDS loads (CDNA5 ISA 7.12.2, wave32) ----
// A (16x32 bf16) row-major [row][K]: lane halves are K = ko..ko+15,
// ko = (laneHi ? 16 : 0)  ->  two contiguous b128 loads.
__device__ __forceinline__ v16bf frag_a_row(const __bf16* rowptr, int lane) {
  const int ko = (lane & 16) ? 16 : 0;
  const v8bf lo = *(const v8bf*)(rowptr + ko);
  const v8bf hi = *(const v8bf*)(rowptr + ko + 8);
  return __builtin_shufflevector(lo, hi, 0,1,2,3,4,5,6,7,8,9,10,11,12,13,14,15);
}
// B (32x16 bf16) stored transposed [N][K] row-major: lane needs K = ko..ko+15 at
// fixed col, ko = (laneHi ? 16 : 0)  ->  one contiguous 32B run = two b128 loads.
__device__ __forceinline__ v16bf frag_b_nk(const __bf16* base, int ld, int lane) {
  const int col = lane & 15;
  const int ko  = (lane & 16) ? 16 : 0;
  const __bf16* p = base + col * ld + ko;
  const v8bf lo = *(const v8bf*)(p);
  const v8bf hi = *(const v8bf*)(p + 8);
  return __builtin_shufflevector(lo, hi, 0,1,2,3,4,5,6,7,8,9,10,11,12,13,14,15);
}

// ---------------- Kernel 1: bias vectors rc = Cn.w_c , rq = Qm.w_q ----------------
__global__ __launch_bounds__(256) void cqa_prep(const float* __restrict__ C,
                                                const float* __restrict__ Q,
                                                const float* __restrict__ W0w,
                                                float* __restrict__ rc,
                                                float* __restrict__ rq) {
  const int b = blockIdx.x;
  __shared__ float w[3 * Dd];
  for (int i = threadIdx.x; i < 3 * Dd; i += 256) w[i] = W0w[i];
  __syncthreads();
  const float* Cb = C + (size_t)b * Dd * Nn;
  const float* Qb = Q + (size_t)b * Dd * Mm;
  for (int n = threadIdx.x; n < Nn; n += 256) {
    float s = 0.f;
#pragma unroll 8
    for (int d = 0; d < Dd; ++d) s += Cb[d * Nn + n] * w[Dd + d];  // w_c
    rc[b * Nn + n] = s;
  }
  for (int m = threadIdx.x; m < Mm; m += 256) {
    float s = 0.f;
#pragma unroll 8
    for (int d = 0; d < Dd; ++d) s += Qb[d * Mm + m] * w[d];       // w_q
    rq[b * Mm + m] = s;
  }
}

// ------- Kernel 2: S = trilinear scores (WMMA) + row softmax + A = S1.Qm (WMMA) -------
// One WG per (batch, 128-row n-tile). 8 waves, each owns 16 n-rows x all 128 m.
// Dynamic LDS: s_cq [n][d] 32KB | s_qt [m][d] 32KB | s_qn [d][m] 32KB  = 96KB.
__global__ __launch_bounds__(256) void cqa_scores(const float* __restrict__ C,
                                                  const float* __restrict__ Q,
                                                  const float* __restrict__ W0w,
                                                  const float* __restrict__ W0b,
                                                  const float* __restrict__ qmask,
                                                  const float* __restrict__ rc,
                                                  const float* __restrict__ rq,
                                                  float* __restrict__ Sws,
                                                  __bf16* __restrict__ S1ws,
                                                  float* __restrict__ Aout) {
  const int b = blockIdx.y, n0 = blockIdx.x * 128;
  const int t = threadIdx.x, lane = t & 31, wv = t >> 5;
  const int colm = lane & 15, hi8 = (lane & 16) ? 8 : 0;
  const int row = wv * 16 + colm;

  extern __shared__ __align__(16) char smem[];
  __bf16* s_cq = (__bf16*)smem;              // (Cn (.) w_cq) [n][d]; reused as S1 strip [n][m]
  __bf16* s_qt = (__bf16*)(smem + 32768);    // Q transposed [m][d]  (B^T for S matmul)
  __bf16* s_qn = (__bf16*)(smem + 65536);    // Q native     [d][m]  (B^T for A matmul)

  const float* Cb = C + (size_t)b * Dd * Nn;
  const float* Qb = Q + (size_t)b * Dd * Mm;

#pragma unroll
  for (int i = 0; i < 16; ++i) {             // 4096 float4 groups
    const int idx4 = i * 256 + t;
    const int dd = idx4 >> 5, mm4 = (idx4 & 31) * 4;
    const v4f qv = *(const v4f*)(Qb + dd * Mm + mm4);
    v4bf qb;
#pragma unroll
    for (int j = 0; j < 4; ++j) qb[j] = to_bf(qv[j]);
    *(v4bf*)(s_qn + dd * Mm + mm4) = qb;               // [d][m], packed b64 store
#pragma unroll
    for (int j = 0; j < 4; ++j) s_qt[(mm4 + j) * Dd + dd] = qb[j];   // [m][d]
    const v4f cv = *(const v4f*)(Cb + dd * Nn + n0 + mm4);
    const float wcq = W0w[2 * Dd + dd];
#pragma unroll
    for (int j = 0; j < 4; ++j) s_cq[(mm4 + j) * Dd + dd] = to_bf(cv[j] * wcq);  // [n][d]
  }
  __syncthreads();

  // ---- S tile: (Cn (.) w_cq) @ Q  -> 16(n) x 128(m) per wave, K=d=128 ----
  v8f acc[8];
#pragma unroll
  for (int i = 0; i < 8; ++i) acc[i] = (v8f){0.f,0.f,0.f,0.f,0.f,0.f,0.f,0.f};
#pragma unroll
  for (int kk = 0; kk < 4; ++kk) {
    const v16bf a = frag_a_row(&s_cq[row * Dd + kk * 32], lane);
#pragma unroll
    for (int mt = 0; mt < 8; ++mt) {
      const v16bf bf = frag_b_nk(&s_qt[(mt * 16) * Dd + kk * 32], Dd, lane);
      acc[mt] = __builtin_amdgcn_wmma_f32_16x16x32_bf16(false, a, false, bf,
                                                        (short)0, acc[mt], false, false);
    }
  }

  // ---- add rank-1 biases + W0_b ----
  const int rbase = n0 + wv * 16 + hi8;
  float rcv[8];
#pragma unroll
  for (int v = 0; v < 8; ++v) rcv[v] = rc[b * Nn + rbase + v];
  const float w0 = W0b[0];
#pragma unroll
  for (int mt = 0; mt < 8; ++mt) {
    const float rqv = rq[b * Mm + mt * 16 + colm];
#pragma unroll
    for (int v = 0; v < 8; ++v) acc[mt][v] += rcv[v] + rqv + w0;
  }

  // ---- store raw S (fp32) for the column softmax ----
  float* Srow = Sws + ((size_t)b * Nn + rbase) * Mm;
#pragma unroll
  for (int mt = 0; mt < 8; ++mt)
#pragma unroll
    for (int v = 0; v < 8; ++v) Srow[v * Mm + mt * 16 + colm] = acc[mt][v];

  // ---- row softmax over m (q_mask applied); each wave owns full rows ----
#pragma unroll
  for (int mt = 0; mt < 8; ++mt) {
    const float pen = BIGF * (1.0f - qmask[b * Mm + mt * 16 + colm]);
#pragma unroll
    for (int v = 0; v < 8; ++v) acc[mt][v] -= pen;
  }
  float inv[8];
#pragma unroll
  for (int v = 0; v < 8; ++v) {
    float mxv = acc[0][v];
#pragma unroll
    for (int mt = 1; mt < 8; ++mt) mxv = fmaxf(mxv, acc[mt][v]);
    for (int s = 1; s < 16; s <<= 1) mxv = fmaxf(mxv, __shfl_xor(mxv, s, 32)); // stays in 16-lane half
    float sm = 0.f;
#pragma unroll
    for (int mt = 0; mt < 8; ++mt) { const float e = __expf(acc[mt][v] - mxv); acc[mt][v] = e; sm += e; }
    for (int s = 1; s < 16; s <<= 1) sm += __shfl_xor(sm, s, 32);
    inv[v] = 1.0f / sm;
  }

  // ---- write S1 (bf16) into this wave's private s_cq strip + global ws ----
  __bf16* s1strip = s_cq + (size_t)(wv * 16) * Mm;
  __bf16* S1g = S1ws + ((size_t)b * Nn + rbase) * Mm;
#pragma unroll
  for (int mt = 0; mt < 8; ++mt)
#pragma unroll
    for (int v = 0; v < 8; ++v) {
      const float p = acc[mt][v] * inv[v];
      const __bf16 pb = to_bf(p);
      s1strip[(v + hi8) * Mm + mt * 16 + colm] = pb;
      S1g[v * Mm + mt * 16 + colm] = pb;
    }

  // ---- A = S1 @ Qm (K=m=128); B^T = Q native [d][m] -> contiguous frag loads ----
  v8f accA[8];
#pragma unroll
  for (int i = 0; i < 8; ++i) accA[i] = (v8f){0.f,0.f,0.f,0.f,0.f,0.f,0.f,0.f};
#pragma unroll
  for (int kk = 0; kk < 4; ++kk) {
    const v16bf a = frag_a_row(&s_cq[row * Mm + kk * 32], lane);
#pragma unroll
    for (int dt = 0; dt < 8; ++dt) {
      const v16bf bf = frag_b_nk(&s_qn[(dt * 16) * Mm + kk * 32], Mm, lane);
      accA[dt] = __builtin_amdgcn_wmma_f32_16x16x32_bf16(false, a, false, bf,
                                                         (short)0, accA[dt], false, false);
    }
  }
  float* Ar = Aout + ((size_t)b * Nn + rbase) * Dd;
#pragma unroll
  for (int dt = 0; dt < 8; ++dt)
#pragma unroll
    for (int v = 0; v < 8; ++v) Ar[v * Dd + dt * 16 + colm] = accA[dt][v];
}

// ---------------- Kernel 3: column softmax over n -> S2 (bf16) ----------------
__global__ __launch_bounds__(256) void cqa_colsm(const float* __restrict__ Sws,
                                                 const float* __restrict__ cmask,
                                                 __bf16* __restrict__ S2ws) {
  const int b = blockIdx.y, m0 = blockIdx.x * 32;
  const int t = threadIdx.x, mx = t & 31, ny = t >> 5;  // 32 columns x 8 n-strides
  __shared__ float red[8][32];
  __shared__ float cred[32];
  const float* Sb = Sws + (size_t)b * Nn * Mm;

  float lmax = -3.0e38f;
  for (int n = ny; n < Nn; n += 8) {
    const float v = Sb[n * Mm + m0 + mx] - BIGF * (1.0f - cmask[b * Nn + n]);
    lmax = fmaxf(lmax, v);
  }
  red[ny][mx] = lmax; __syncthreads();
  if (ny == 0) { float m = red[0][mx];
#pragma unroll
    for (int i = 1; i < 8; ++i) m = fmaxf(m, red[i][mx]);
    cred[mx] = m; }
  __syncthreads();
  const float cm = cred[mx];
  __syncthreads();

  float lsum = 0.f;
  for (int n = ny; n < Nn; n += 8) {
    const float v = Sb[n * Mm + m0 + mx] - BIGF * (1.0f - cmask[b * Nn + n]);
    lsum += __expf(v - cm);
  }
  red[ny][mx] = lsum; __syncthreads();
  if (ny == 0) { float s = red[0][mx];
#pragma unroll
    for (int i = 1; i < 8; ++i) s += red[i][mx];
    cred[mx] = s; }
  __syncthreads();
  const float invs = 1.0f / cred[mx];

  __bf16* S2b = S2ws + (size_t)b * Nn * Mm;
  for (int n = ny; n < Nn; n += 8) {
    const float v = Sb[n * Mm + m0 + mx] - BIGF * (1.0f - cmask[b * Nn + n]);
    S2b[n * Mm + m0 + mx] = to_bf(__expf(v - cm) * invs);
  }
}

// ---------------- Kernel 4: T-partials = S2^T @ Cn (split-K over n by 4) ----------------
// Dynamic LDS: sa_t [m(128)][n(32)] 8KB | sb [d(128)][n(32)] 8KB.
__global__ __launch_bounds__(256) void cqa_t(const float* __restrict__ C,
                                             const __bf16* __restrict__ S2ws,
                                             float* __restrict__ Tp) {
  const int b = blockIdx.y, ch = blockIdx.x;  // n-chunk 0..3 (256 rows each)
  const int t = threadIdx.x, lane = t & 31, wv = t >> 5;
  const int colm = lane & 15, hi8 = (lane & 16) ? 8 : 0;
  const int rowm = wv * 16 + colm;            // output m-row

  extern __shared__ __align__(16) char smem[];
  __bf16* sa_t = (__bf16*)smem;               // S2^T chunk [m][n32] (A row-major)
  __bf16* sb   = (__bf16*)(smem + 8192);      // C chunk [d][n32]   (B^T = [N][K])

  const float*  Cb  = C + (size_t)b * Dd * Nn;
  const __bf16* S2b = S2ws + (size_t)b * Nn * Mm;

  v8f acc[8];
#pragma unroll
  for (int i = 0; i < 8; ++i) acc[i] = (v8f){0.f,0.f,0.f,0.f,0.f,0.f,0.f,0.f};

  for (int ks = 0; ks < 8; ++ks) {            // K = 256 per chunk, 32 per step
    const int nb = ch * 256 + ks * 32;
    __syncthreads();
#pragma unroll
    for (int i = 0; i < 2; ++i) {             // 512 bf16x8 groups (coalesced b128 reads)
      const int idx8 = i * 256 + t;
      const int nn = idx8 >> 4, mm8 = (idx8 & 15) * 8;
      const v8bf sv = *(const v8bf*)(S2b + (size_t)(nb + nn) * Mm + mm8);
#pragma unroll
      for (int j = 0; j < 8; ++j) sa_t[(mm8 + j) * 32 + nn] = sv[j];   // LDS transpose scatter
    }
#pragma unroll
    for (int i = 0; i < 4; ++i) {             // 1024 float4 groups
      const int idx4 = i * 256 + t;
      const int dd = idx4 >> 3, nn4 = (idx4 & 7) * 4;
      const v4f cv = *(const v4f*)(Cb + (size_t)dd * Nn + nb + nn4);
      v4bf cb;
#pragma unroll
      for (int j = 0; j < 4; ++j) cb[j] = to_bf(cv[j]);
      *(v4bf*)(sb + dd * 32 + nn4) = cb;      // packed b64 store
    }
    __syncthreads();
    const v16bf a = frag_a_row(&sa_t[rowm * 32], lane);    // A[m][k=n]
#pragma unroll
    for (int dt = 0; dt < 8; ++dt) {
      const v16bf bf = frag_b_nk(&sb[(dt * 16) * 32], 32, lane);  // B[k=n][d] = C[d][n]
      acc[dt] = __builtin_amdgcn_wmma_f32_16x16x32_bf16(false, a, false, bf,
                                                        (short)0, acc[dt], false, false);
    }
  }
  float* Tq = Tp + (((size_t)b * 4 + ch) * Mm + wv * 16 + hi8) * Dd;
#pragma unroll
  for (int dt = 0; dt < 8; ++dt)
#pragma unroll
    for (int v = 0; v < 8; ++v) Tq[v * Dd + dt * 16 + colm] = acc[dt][v];
}

// ---------------- Kernel 5: B = S1 @ T  (sums T-partials in LDS) ----------------
// Dynamic LDS: s1 [n][m] 32KB | st_t [d][m] 32KB.
// S1 tile is a pure 32KB global->LDS copy: use the CDNA5 async-to-LDS path when
// available and overlap it with the T-partial summation VALU work.
__global__ __launch_bounds__(256) void cqa_b(const __bf16* __restrict__ S1ws,
                                             const float* __restrict__ Tp,
                                             float* __restrict__ Bout) {
  const int b = blockIdx.y, n0 = blockIdx.x * 128;
  const int t = threadIdx.x, lane = t & 31, wv = t >> 5;
  const int colm = lane & 15, hi8 = (lane & 16) ? 8 : 0;
  const int row = wv * 16 + colm;

  extern __shared__ __align__(16) char smem[];
  __bf16* s1   = (__bf16*)smem;               // S1 tile [n][m] (A row-major)
  __bf16* st_t = (__bf16*)(smem + 32768);     // T^T [d][m]     (B^T = [N][K])

  const __bf16* S1b = S1ws + ((size_t)b * Nn + n0) * Mm;
  const float*  Tb  = Tp + (size_t)b * 4 * Mm * Dd;

#if __has_builtin(__builtin_amdgcn_global_load_async_to_lds_b128)
  {
    // Generic->AS casts via integers: global generic == AS1 bit-identical;
    // generic LDS pointer's low 32 bits are the LDS byte offset (ISA aperture rule).
    GLOBAL_AS v4i* gsrc = (GLOBAL_AS v4i*)(uintptr_t)S1b;
    LDS_AS   v4i* ldst  = (LDS_AS   v4i*)(unsigned)(uintptr_t)s1;
#pragma unroll
    for (int i = 0; i < 8; ++i) {             // 32KB, 16B per lane per issue
      const int vidx = i * 256 + t;
      __builtin_amdgcn_global_load_async_to_lds_b128(gsrc + vidx, ldst + vidx, 0, 0);
    }
  }
#else
#pragma unroll
  for (int i = 0; i < 8; ++i) {
    const int idx8 = i * 256 + t;
    *(v8bf*)(s1 + idx8 * 8) = *(const v8bf*)(S1b + (size_t)idx8 * 8);
  }
#endif

  // st_t: sum 4 split-K partials (b128 reads), transpose to [d][m] while converting
#pragma unroll
  for (int i = 0; i < 16; ++i) {
    const int idx4 = i * 256 + t;
    const int mm = idx4 >> 5, dd4 = (idx4 & 31) * 4;
    const size_t o = (size_t)mm * Dd + dd4;
    v4f v = *(const v4f*)(Tb + o);
    v = v + *(const v4f*)(Tb + (size_t)Mm * Dd + o);
    v = v + *(const v4f*)(Tb + (size_t)2 * Mm * Dd + o);
    v = v + *(const v4f*)(Tb + (size_t)3 * Mm * Dd + o);
#pragma unroll
    for (int j = 0; j < 4; ++j) st_t[(dd4 + j) * Mm + mm] = to_bf(v[j]);
  }

#if __has_builtin(__builtin_amdgcn_global_load_async_to_lds_b128)
#if __has_builtin(__builtin_amdgcn_s_wait_asynccnt)
  __builtin_amdgcn_s_wait_asynccnt(0);
#else
  asm volatile("s_wait_asynccnt 0x0" ::: "memory");
#endif
#endif
  __syncthreads();

  v8f acc[8];
#pragma unroll
  for (int i = 0; i < 8; ++i) acc[i] = (v8f){0.f,0.f,0.f,0.f,0.f,0.f,0.f,0.f};
#pragma unroll
  for (int kk = 0; kk < 4; ++kk) {            // K = m = 128
    const v16bf a = frag_a_row(&s1[row * Mm + kk * 32], lane);
#pragma unroll
    for (int dt = 0; dt < 8; ++dt) {
      const v16bf bf = frag_b_nk(&st_t[(dt * 16) * Mm + kk * 32], Mm, lane);
      acc[dt] = __builtin_amdgcn_wmma_f32_16x16x32_bf16(false, a, false, bf,
                                                        (short)0, acc[dt], false, false);
    }
  }
  float* Br = Bout + ((size_t)b * Nn + n0 + wv * 16 + hi8) * Dd;
#pragma unroll
  for (int dt = 0; dt < 8; ++dt)
#pragma unroll
    for (int v = 0; v < 8; ++v) Br[v * Dd + dt * 16 + colm] = acc[dt][v];
}

extern "C" void kernel_launch(void* const* d_in, const int* in_sizes, int n_in,
                              void* d_out, int out_size, void* d_ws, size_t ws_size,
                              hipStream_t stream) {
  (void)in_sizes; (void)n_in; (void)out_size; (void)ws_size;
  const float* C     = (const float*)d_in[0];
  const float* Q     = (const float*)d_in[1];
  const float* cmask = (const float*)d_in[2];
  const float* qmask = (const float*)d_in[3];
  const float* W0w   = (const float*)d_in[4];
  const float* W0b   = (const float*)d_in[5];

  float* Aout = (float*)d_out;
  float* Bout = (float*)d_out + (size_t)Bb * Nn * Dd;

  // workspace layout (~83 MB)
  char* ws = (char*)d_ws;
  const size_t SZ_S = (size_t)Bb * Nn * Mm;           // 8.39M elems
  float*  Sws  = (float*)ws;                          // raw S, fp32
  __bf16* S1ws = (__bf16*)(ws + SZ_S * 4);            // S1, bf16
  __bf16* S2ws = (__bf16*)(ws + SZ_S * 4 + SZ_S * 2); // S2, bf16
  float*  Tp   = (float*)(ws + SZ_S * 4 + SZ_S * 4);  // 4 split-K partials of T
  float*  rc   = (float*)(ws + SZ_S * 4 + SZ_S * 4 + (size_t)Bb * 4 * Mm * Dd * 4);
  float*  rq   = rc + (size_t)Bb * Nn;

  cqa_prep  <<<Bb,               256, 0,     stream>>>(C, Q, W0w, rc, rq);
  cqa_scores<<<dim3(Nn/128, Bb), 256, 98304, stream>>>(C, Q, W0w, W0b, qmask, rc, rq, Sws, S1ws, Aout);
  cqa_colsm <<<dim3(Mm/32,  Bb), 256, 0,     stream>>>(Sws, cmask, S2ws);
  cqa_t     <<<dim3(4,      Bb), 256, 16384, stream>>>(C, S2ws, Tp);
  cqa_b     <<<dim3(Nn/128, Bb), 256, 65536, stream>>>(S1ws, Tp, Bout);
}
